// multiTimeAttention_43508018708876
// MI455X (gfx1250) — compile-verified, matrix operand order checked by reading, not run on hardware
//
#include <hip/hip_runtime.h>
#include <hip/hip_bf16.h>
#include <hip/hip_fp16.h>

// ---------------------------------------------------------------------------
// multiTimeAttention for MI455X (gfx1250, wave32, WMMA + TDM)
//
// Algebraic restructuring:
//   Wk_f = Wk@Wkc, bk_f = Wk@bkc+bk   (1x1-conv -> linear chain collapsed)
//   scores contract kp summed over v  ->  kst = (sum_v k) @ Wk_f^T + 64*bk_f
//   (kp is NEVER materialized; saves 34 GFLOP + 512MB traffic)
//   vp  = v @ Wv_f^T + bv_f            (stored transposed f16 as GEMM-B)
//   S   = qp @ kst^T * 1/sqrt(nk*nv);  P = softmax(S)
//   O   = P @ vpT                      (batched per (b,h))
//   out = O @ Wo^T + bo
// Heavy GEMMs: v_wmma_f32_16x16x32_f16, B (and f16-A) tiles staged into LDS
// by the Tensor Data Mover (tensor_load_to_lds + s_wait_tensorcnt) with
// descriptor-driven padding to a conflict-free 40-half row stride.
// ---------------------------------------------------------------------------

typedef __attribute__((ext_vector_type(16))) _Float16 v16h;
typedef __attribute__((ext_vector_type(8)))  _Float16 v8h;
typedef __attribute__((ext_vector_type(8)))  float    v8f;
typedef __attribute__((ext_vector_type(4)))  unsigned v4u;
typedef __attribute__((ext_vector_type(8)))  int      v8i;
typedef __attribute__((ext_vector_type(4)))  int      v4i;

#define EMBED 256
#define NHID  512
#define NHEAD 8
#define HD    64
#define BATCH 4
#define NQ    512
#define NK    512
#define NV    64

// ---- TDM availability / arity guards (probe-verified per CDNA5_HIP.md) ----
#if defined(__has_builtin)
#if __has_builtin(__builtin_amdgcn_tensor_load_to_lds) && \
    __has_builtin(__builtin_amdgcn_s_wait_tensorcnt)
#define USE_TDM 1
#endif
#endif
#ifndef USE_TDM
#define USE_TDM 0
#endif

#if USE_TDM
#if __clang_major__ >= 23
#define TDM_CALL(g0, g1, g2, g3) \
  __builtin_amdgcn_tensor_load_to_lds((g0), (g1), (g2), (g3), (v8i){0}, 0)
#else
#define TDM_CALL(g0, g1, g2, g3) \
  __builtin_amdgcn_tensor_load_to_lds((g0), (g1), (g2), (g3), 0)
#endif

// 2D f16 tile load: rows x 32 elements, row stride K elements, into LDS at
// ldsOff with 16B padding after every 64B row -> 40-half LDS row stride.
__device__ __forceinline__ void tdm_load_tile(unsigned ldsOff,
                                              const _Float16* gptr, int rows,
                                              int K) {
  const unsigned long long ga = (unsigned long long)(size_t)gptr;
  // group0: count=1 | lds_addr | global_addr[56:0] | type=2
  v4u g0 = {1u, ldsOff, (unsigned)ga, (unsigned)(ga >> 32) | (2u << 30)};
  // group1: data_size=2B, pad_enable, pad_interval=16 DW, pad_amount=4 DW,
  //         tensor_dim0=32, tensor_dim1=rows, tile_dim0=32, tile_dim1=rows,
  //         tensor_dim0_stride=K
  v8i g1 = {(int)((1u << 16) | (1u << 20) | (3u << 22) | (3u << 25)),
            (int)(32u << 16),
            (int)((unsigned)rows << 16),
            (int)(32u << 16),
            rows,
            K,
            0,
            0};
  v4i gz = {0, 0, 0, 0};
  TDM_CALL(g0, g1, gz, gz);
}
#endif  // USE_TDM

// ------------------------- tiny prep kernels -------------------------------

__global__ void fuse_weights_kernel(const float* __restrict__ W,
                                    const float* __restrict__ Wc,
                                    const float* __restrict__ bc,
                                    const float* __restrict__ b,
                                    _Float16* __restrict__ Wf,
                                    float* __restrict__ bf, float biasScale) {
  const int n = blockIdx.x;   // 0..511
  const int c = threadIdx.x;  // 0..255
  float acc = 0.f;
  for (int e = 0; e < EMBED; ++e)
    acc += W[n * EMBED + e] * Wc[e * EMBED + c];
  Wf[n * EMBED + c] = (_Float16)acc;
  if (c == 0) {
    float ab = 0.f;
    for (int e = 0; e < EMBED; ++e) ab += W[n * EMBED + e] * bc[e];
    bf[n] = biasScale * (ab + b[n]);
  }
}

__global__ void f32_to_f16_kernel(const float* __restrict__ src,
                                  _Float16* __restrict__ dst, int n) {
  int i = blockIdx.x * blockDim.x + threadIdx.x;
  if (i < n) dst[i] = (_Float16)src[i];
}

// out[b*512+kk][e] = sum_v k[b][kk][v][e]
__global__ void ksum_kernel(const float* __restrict__ kin,
                            _Float16* __restrict__ out) {
  const int e = threadIdx.x;   // 0..255
  const int bk = blockIdx.x;   // 0..2047
  const float* p = kin + (size_t)bk * NV * EMBED + e;
  float s = 0.f;
#pragma unroll 8
  for (int v = 0; v < NV; ++v) s += p[(size_t)v * EMBED];
  out[(size_t)bk * EMBED + e] = (_Float16)s;
}

// ------------------------- generic WMMA GEMM -------------------------------
// C[m,n] = sum_k A[m][k]*B[n][k] + bias[n]   (A: [M][K], B: [N][K] row-major)
// Block tile 64x256, BK=32; 256 threads = 8 waves (2x4); 32x64 per wave
// (2x4 fragments -> 8 WMMA per wave per K-step).

enum { OUT_QP = 0, OUT_VPT = 1, OUT_OBUF = 2, OUT_F32 = 3 };

template <typename AT, int MODE>
__global__ __launch_bounds__(256) void gemm_wmma_kernel(
    const AT* __restrict__ A, const _Float16* __restrict__ Bw,
    const float* __restrict__ bias, void* __restrict__ outp, int M, int N,
    int K, long aBatchStride, long bBatchStride) {
  constexpr int BM = 64, BN = 256, BK = 32, LDA = 40, LDB = 40;
  __shared__ __align__(16) _Float16 As[BM * LDA];  // 5 KB
  __shared__ __align__(16) _Float16 Bs[BN * LDB];  // 20 KB

  const int t = threadIdx.x;
  const int lane = t & 31;
  const int wave = t >> 5;
  const int wm = wave & 1;   // M sub-tile of 32
  const int wn = wave >> 1;  // N sub-tile of 64
  const int m0 = blockIdx.x * BM;
  const int n0 = blockIdx.y * BN;
  const int z  = blockIdx.z;

  const AT* Ab = A + (size_t)z * aBatchStride;
  const _Float16* Bb = Bw + (size_t)z * bBatchStride;

  v8f acc[2][4] = {};

  const int arow = t >> 2, acol = (t & 3) * 8;  // A tile: 8 elems/thread
  const int fr = lane & 15;                     // fragment row (M or N)
  const int k0 = (lane >> 4) << 3;              // 0 or 8 (lane-half K offset)

#if USE_TDM
  const unsigned ldsAoff = (unsigned)(size_t)(const void*)As;
  const unsigned ldsBoff = (unsigned)(size_t)(const void*)Bs;
#endif

  for (int kk = 0; kk < K; kk += BK) {
#if USE_TDM
    if (t < 32) {  // wave 0 drives the Tensor Data Mover
      tdm_load_tile(ldsBoff, Bb + (size_t)n0 * K + kk, BN, K);
      if constexpr (sizeof(AT) == 2)
        tdm_load_tile(ldsAoff,
                      (const _Float16*)(const void*)Ab + (size_t)m0 * K + kk,
                      BM, K);
    }
    if constexpr (sizeof(AT) != 2) {  // f32 A: convert on the LDS path
      const AT* ap = Ab + (size_t)(m0 + arow) * K + kk + acol;
#pragma unroll
      for (int j = 0; j < 8; ++j) As[arow * LDA + acol + j] = (_Float16)ap[j];
      if (kk + BK < K)
        __builtin_prefetch(Ab + (size_t)(m0 + arow) * K + kk + BK + acol, 0, 3);
    }
    if (t < 32) __builtin_amdgcn_s_wait_tensorcnt(0);
#else
    {
      const AT* ap = Ab + (size_t)(m0 + arow) * K + kk + acol;
#pragma unroll
      for (int j = 0; j < 8; ++j) As[arow * LDA + acol + j] = (_Float16)ap[j];
      const _Float16* bp = Bb + (size_t)(n0 + t) * K + kk;
#pragma unroll
      for (int j = 0; j < 32; ++j) Bs[t * LDB + j] = bp[j];
      if (kk + BK < K) {
        __builtin_prefetch(Ab + (size_t)(m0 + arow) * K + kk + BK + acol, 0, 3);
        __builtin_prefetch(Bb + (size_t)(n0 + t) * K + kk + BK, 0, 3);
      }
    }
#endif
    __syncthreads();

    // fragment loads per 16-bit A/B VGPR layouts (b128 LDS reads)
    v16h afrag[2], bfrag[4];
#pragma unroll
    for (int mi = 0; mi < 2; ++mi) {
      const _Float16* pa = &As[(wm * 32 + mi * 16 + fr) * LDA];
      const v8h c0 = *(const v8h*)(pa + k0);
      const v8h c1 = *(const v8h*)(pa + 16 + k0);
#pragma unroll
      for (int j = 0; j < 8; ++j) {
        afrag[mi][j] = c0[j];
        afrag[mi][8 + j] = c1[j];
      }
    }
#pragma unroll
    for (int ni = 0; ni < 4; ++ni) {
      const _Float16* pb = &Bs[(wn * 64 + ni * 16 + fr) * LDB];
      const v8h c0 = *(const v8h*)(pb + k0);
      const v8h c1 = *(const v8h*)(pb + 16 + k0);
#pragma unroll
      for (int j = 0; j < 8; ++j) {
        bfrag[ni][j] = c0[j];
        bfrag[ni][8 + j] = c1[j];
      }
    }
#pragma unroll
    for (int mi = 0; mi < 2; ++mi)
#pragma unroll
      for (int ni = 0; ni < 4; ++ni)
        acc[mi][ni] = __builtin_amdgcn_wmma_f32_16x16x32_f16(
            false, afrag[mi], false, bfrag[ni], (short)0, acc[mi][ni], false,
            false);
    __syncthreads();
  }

  // epilogue: bias + scatter per 16x16 f32 C/D layout
  const int r0 = (lane < 16) ? 0 : 8;
  const int cn = lane & 15;
#pragma unroll
  for (int mi = 0; mi < 2; ++mi) {
#pragma unroll
    for (int ni = 0; ni < 4; ++ni) {
      const int gn = n0 + wn * 64 + ni * 16 + cn;
      const float bv = bias ? bias[gn] : 0.f;
#pragma unroll
      for (int i = 0; i < 8; ++i) {
        const int gm = m0 + wm * 32 + mi * 16 + r0 + i;
        const float val = acc[mi][ni][i] + bv;
        if (MODE == OUT_QP) {  // [b][h][q][d]  (m=(b,q), n=(h,d))
          const int b = gm >> 9, q = gm & 511, h = gn >> 6, d = gn & 63;
          ((_Float16*)outp)[((((size_t)b * NHEAD + h) * NQ + q) << 6) + d] =
              (_Float16)val;
        } else if (MODE == OUT_VPT) {  // [b][h][(v,d)][k] (m=(b,k,v), n=(h,d))
          const int b = gm >> 15, kx = (gm >> 6) & 511, vv = gm & 63;
          const int h = gn >> 6, d = gn & 63;
          ((_Float16*)outp)[((size_t)(b * NHEAD + h) * 4096 + vv * 64 + d) *
                                NK + kx] = (_Float16)val;
        } else if (MODE == OUT_OBUF) {  // [(b,q,v)][(h,d)] (z=(b,h), n=(v,d))
          const int b = z >> 3, h = z & 7;
          const int vv = gn >> 6, d = gn & 63;
          ((_Float16*)outp)[((size_t)(b * NQ + gm) * NV + vv) * NHID + h * HD +
                            d] = (_Float16)val;
        } else {  // OUT_F32: plain row-major f32
          ((float*)outp)[(size_t)gm * N + gn] = val;
        }
      }
    }
  }
}

// -------------------- scores + softmax (fused, WMMA) -----------------------
__global__ __launch_bounds__(512) void attn_softmax_kernel(
    const _Float16* __restrict__ qp,   // [32][512][64]
    const _Float16* __restrict__ kst,  // [32][512][64]
    _Float16* __restrict__ P) {        // [32][512][512]
  __shared__ float S[16][NK + 4];
  __shared__ float rowred[16];
  __shared__ float partial[16][32];

  const int bh = blockIdx.y;
  const int q0 = blockIdx.x * 16;
  const int t = threadIdx.x;
  const int lane = t & 31;
  const int wave = t >> 5;  // 0..15
  const float scale = 0.0055242717280199026f;  // 1/sqrt(512*64)

  const int fr = lane & 15;
  const int k0 = (lane >> 4) << 3;

  v16h a0, a1;
  {
    const _Float16* pr = qp + ((size_t)bh * NQ + q0 + fr) * HD;
    const v8h c0 = *(const v8h*)(pr + k0);
    const v8h c1 = *(const v8h*)(pr + 16 + k0);
    const v8h c2 = *(const v8h*)(pr + 32 + k0);
    const v8h c3 = *(const v8h*)(pr + 48 + k0);
#pragma unroll
    for (int j = 0; j < 8; ++j) {
      a0[j] = c0[j]; a0[8 + j] = c1[j];
      a1[j] = c2[j]; a1[8 + j] = c3[j];
    }
  }
  const _Float16* kb = kst + (size_t)bh * NK * HD;
#pragma unroll
  for (int ti = 0; ti < 2; ++ti) {
    const int n0 = wave * 32 + ti * 16;
    v16h b0, b1;
    {
      const _Float16* pr = kb + (size_t)(n0 + fr) * HD;
      const v8h c0 = *(const v8h*)(pr + k0);
      const v8h c1 = *(const v8h*)(pr + 16 + k0);
      const v8h c2 = *(const v8h*)(pr + 32 + k0);
      const v8h c3 = *(const v8h*)(pr + 48 + k0);
#pragma unroll
      for (int j = 0; j < 8; ++j) {
        b0[j] = c0[j]; b0[8 + j] = c1[j];
        b1[j] = c2[j]; b1[8 + j] = c3[j];
      }
    }
    v8f acc = {};
    acc = __builtin_amdgcn_wmma_f32_16x16x32_f16(false, a0, false, b0,
                                                 (short)0, acc, false, false);
    acc = __builtin_amdgcn_wmma_f32_16x16x32_f16(false, a1, false, b1,
                                                 (short)0, acc, false, false);
    const int r0 = (lane < 16) ? 0 : 8;
    const int cn = n0 + (lane & 15);
#pragma unroll
    for (int i = 0; i < 8; ++i) S[r0 + i][cn] = acc[i] * scale;
  }
  __syncthreads();

  if (t < 16) {
    float m = -1e30f;
    for (int c = 0; c < NK; ++c) m = fmaxf(m, S[t][c]);
    rowred[t] = m;
  }
  __syncthreads();
  {
    const int row = t >> 5, seg = t & 31;
    const float rm = rowred[row];
    float s = 0.f;
#pragma unroll
    for (int j = 0; j < 16; ++j) {
      const float e = __expf(S[row][seg * 16 + j] - rm);
      S[row][seg * 16 + j] = e;
      s += e;
    }
    partial[row][seg] = s;
  }
  __syncthreads();
  if (t < 16) {
    float s = 0.f;
    for (int j = 0; j < 32; ++j) s += partial[t][j];
    rowred[t] = 1.f / s;
  }
  __syncthreads();
  {
    const int row = t >> 5, seg = t & 31;
    const float inv = rowred[row];
    _Float16* pp = P + ((size_t)bh * NQ + q0 + row) * NK + seg * 16;
#pragma unroll
    for (int j = 0; j < 16; ++j)
      pp[j] = (_Float16)(S[row][seg * 16 + j] * inv);
  }
}

// ------------------------------ launcher -----------------------------------

extern "C" void kernel_launch(void* const* d_in, const int* in_sizes, int n_in,
                              void* d_out, int out_size, void* d_ws,
                              size_t ws_size, hipStream_t stream) {
  (void)in_sizes; (void)n_in; (void)out_size; (void)ws_size;

  const float* q   = (const float*)d_in[0];
  const float* kin = (const float*)d_in[1];
  const float* vin = (const float*)d_in[2];
  const float* Wq  = (const float*)d_in[3];
  const float* bq  = (const float*)d_in[4];
  const float* Wk  = (const float*)d_in[5];
  const float* bk  = (const float*)d_in[6];
  const float* Wv  = (const float*)d_in[7];
  const float* bv  = (const float*)d_in[8];
  const float* Wo  = (const float*)d_in[9];
  const float* bo  = (const float*)d_in[10];
  const float* Wkc = (const float*)d_in[11];
  const float* bkc = (const float*)d_in[12];
  const float* Wvc = (const float*)d_in[13];
  const float* bvc = (const float*)d_in[14];

  // workspace map (all 256B-aligned; total ~292 MB)
  char* ws = (char*)d_ws;
  _Float16* Wkf   = (_Float16*)(ws + 0);
  _Float16* Wvf   = (_Float16*)(ws + 262144);
  _Float16* Wq16  = (_Float16*)(ws + 524288);
  _Float16* Wo16  = (_Float16*)(ws + 786432);
  float*    biasK = (float*)   (ws + 1048576);    // 64*(Wk@bkc+bk)
  float*    biasV = (float*)   (ws + 1050624);
  _Float16* qp16  = (_Float16*)(ws + 1052672);    // [32][512][64]
  _Float16* kst16 = (_Float16*)(ws + 3149824);    // [32][512][64]
  _Float16* ksum  = (_Float16*)(ws + 5246976);    // [2048][256]
  _Float16* P16   = (_Float16*)(ws + 6295552);    // [32][512][512]
  _Float16* vpT   = (_Float16*)(ws + 23072768);   // [32][4096][512]
  _Float16* O16   = (_Float16*)(ws + 157290496);  // [131072][512]

  fuse_weights_kernel<<<dim3(512), 256, 0, stream>>>(Wk, Wkc, bkc, bk, Wkf,
                                                     biasK, 64.f);
  fuse_weights_kernel<<<dim3(512), 256, 0, stream>>>(Wv, Wvc, bvc, bv, Wvf,
                                                     biasV, 1.f);
  f32_to_f16_kernel<<<dim3(512), 256, 0, stream>>>(Wq, Wq16, NHID * EMBED);
  f32_to_f16_kernel<<<dim3(512), 256, 0, stream>>>(Wo, Wo16, EMBED * NHID);
  ksum_kernel<<<dim3(2048), 256, 0, stream>>>(kin, ksum);

  // qp = q@Wq^T+bq   M=2048 N=512 K=256
  gemm_wmma_kernel<float, OUT_QP><<<dim3(32, 2, 1), 256, 0, stream>>>(
      q, Wq16, bq, qp16, 2048, 512, 256, 0, 0);
  // kst = ksum@Wkf^T + 64*bkf
  gemm_wmma_kernel<_Float16, OUT_QP><<<dim3(32, 2, 1), 256, 0, stream>>>(
      ksum, Wkf, biasK, kst16, 2048, 512, 256, 0, 0);
  // vpT = (v@Wvf^T+bvf) -> [b][h][(v,d)][k]   M=131072 N=512 K=256
  gemm_wmma_kernel<float, OUT_VPT><<<dim3(2048, 2, 1), 256, 0, stream>>>(
      vin, Wvf, biasV, vpT, 131072, 512, 256, 0, 0);
  // P = softmax(scale * qp@kst^T)
  attn_softmax_kernel<<<dim3(32, 32), 512, 0, stream>>>(qp16, kst16, P16);
  // O = P@vpT per (b,h): M=512 N=4096 K=512 -> [(b,q,v)][(h,d)]
  gemm_wmma_kernel<_Float16, OUT_OBUF><<<dim3(8, 16, 32), 256, 0, stream>>>(
      P16, vpT, nullptr, O16, 512, 4096, 512, (long)NQ * NK, (long)4096 * NK);
  // out = O@Wo^T + bo (f32): M=131072 N=256 K=512
  gemm_wmma_kernel<_Float16, OUT_F32><<<dim3(2048, 1, 1), 256, 0, stream>>>(
      O16, Wo16, bo, d_out, 131072, 256, 512, 0, 0);
}
